// GNNAgent_53815940219242
// MI455X (gfx1250) — matrix operand order repaired and checked by reading, not compiled, not run on hardware
//
#include <hip/hip_runtime.h>
#include <hip/hip_bf16.h>
#include <stdint.h>

#define NN   50000
#define NE   640000
#define DH   128
#define OUTD 32

typedef __attribute__((ext_vector_type(16))) __bf16 v16bf;
typedef __attribute__((ext_vector_type(8)))  float  v8f;
typedef __attribute__((ext_vector_type(4)))  int    v4i;

// hardware f32 -> bf16 (RNE) convert
__device__ __forceinline__ uint16_t f2bf(float f) {
  union { __bf16 h; uint16_t u; } c;
  c.h = (__bf16)f;
  return c.u;
}

union FragU { uint4 u[2]; v16bf v; };

// A-fragment (16x32 bf16, row-major source):
// lane<16: M=lane, K = kb+{0..7} and kb+16+{0..7}
// lane>=16: M=lane-16, K = kb+8+{0..7} and kb+24+{0..7}
__device__ __forceinline__ v16bf load_afrag(const uint16_t* row, int kb, int hi) {
  FragU f;
  f.u[0] = *(const uint4*)(row + kb + hi * 8);
  f.u[1] = *(const uint4*)(row + kb + 16 + hi * 8);
  return f.v;
}

// B-fragment (32x16 bf16) from transposed weights stored [n][k] (works for LDS or global ptr):
// lane<16: N=lane, K = kb+{0..7} and kb+16+{0..7}; lane>=16: N=lane-16, K = kb+8+{0..7} and kb+24+{0..7}
__device__ __forceinline__ v16bf load_bfrag(const uint16_t* row, int kb, int hi) {
  FragU f;
  const uint16_t* p = row + kb + hi * 8;
  f.u[0] = *(const uint4*)(p);
  f.u[1] = *(const uint4*)(p + 16);
  return f.v;
}

#define WMMA_BF16(a, b, c) \
  __builtin_amdgcn_wmma_f32_16x16x32_bf16(false, (a), false, (b), (short)0, (c), false, false)

// ---- stage a weight matrix into LDS (blockDim.x == 256) ----
#if __has_builtin(__builtin_amdgcn_global_load_async_to_lds_b128) && \
    __has_builtin(__builtin_amdgcn_s_wait_asynccnt)
#define HAVE_ASYNC_LDS 1
#else
#define HAVE_ASYNC_LDS 0
#endif

__device__ __forceinline__ void stage_to_lds(uint16_t* smem, const uint16_t* gmem,
                                             int total_bytes) {
  const int n16 = total_bytes >> 4;   // # of 16B chunks, multiple of 256
#if HAVE_ASYNC_LDS
  for (int i = threadIdx.x; i < n16; i += 256) {
    auto dst = (__attribute__((address_space(3))) v4i*)((char*)smem + i * 16);
    auto src = (__attribute__((address_space(1))) v4i*)((char*)gmem + i * 16);
    __builtin_amdgcn_global_load_async_to_lds_b128(src, dst, 0, 0);
  }
  __builtin_amdgcn_s_wait_asynccnt(0);
#else
  for (int i = threadIdx.x; i < n16; i += 256) {
    ((uint4*)smem)[i] = ((const uint4*)gmem)[i];
  }
#endif
  __syncthreads();
}

// ---------------- prep kernels ----------------

__global__ void k_cvt_x(const float* __restrict__ x, uint16_t* __restrict__ xbf, int n) {
  int i = blockIdx.x * blockDim.x + threadIdx.x;
  if (i < n) xbf[i] = f2bf(x[i]);
}

// W: [K][Ncol] row-major f32  ->  WT: [Ncol][K] bf16
__global__ void k_transpose_w(const float* __restrict__ W, uint16_t* __restrict__ WT,
                              int K, int Ncol) {
  int i = blockIdx.x * blockDim.x + threadIdx.x;
  if (i < K * Ncol) {
    int k = i / Ncol, n = i % Ncol;
    WT[n * K + k] = f2bf(W[i]);
  }
}

// ---------------- h0 = x @ Wp + bp ----------------

__global__ void __launch_bounds__(256)
k_proj(const uint16_t* __restrict__ xbf, const uint16_t* __restrict__ WpT,
       const float* __restrict__ bp,
       float* __restrict__ h0f, uint16_t* __restrict__ h0bf) {
  __shared__ uint16_t sW[128 * 128];          // 32 KB
  stage_to_lds(sW, WpT, 128 * 128 * 2);

  const int wave = (blockIdx.x * blockDim.x + threadIdx.x) >> 5;
  const int lane = threadIdx.x & 31;
  const int ntiles = NN / 16;
  if (wave >= ntiles) return;                 // wave-uniform: EXEC stays all-ones
  const int row0 = wave * 16;
  const int hi = lane >> 4;
  const int ln = lane & 15;

  const uint16_t* arow = xbf + (size_t)(row0 + ln) * DH;
  v16bf a0 = load_afrag(arow, 0, hi);
  v16bf a1 = load_afrag(arow, 32, hi);
  v16bf a2 = load_afrag(arow, 64, hi);
  v16bf a3 = load_afrag(arow, 96, hi);

  for (int ct = 0; ct < 8; ++ct) {
    const int col0 = ct * 16;
    const uint16_t* brow = sW + (size_t)(col0 + ln) * DH;   // LDS
    float bias = bp[col0 + ln];
    v8f c = {bias, bias, bias, bias, bias, bias, bias, bias};
    c = WMMA_BF16(a0, load_bfrag(brow, 0, hi), c);
    c = WMMA_BF16(a1, load_bfrag(brow, 32, hi), c);
    c = WMMA_BF16(a2, load_bfrag(brow, 64, hi), c);
    c = WMMA_BF16(a3, load_bfrag(brow, 96, hi), c);
#pragma unroll
    for (int r = 0; r < 8; ++r) {             // D layout: lane=col, VGPR r -> row r+8*hi
      int m = r + hi * 8;
      float v = c[r];
      size_t idx = (size_t)(row0 + m) * DH + col0 + ln;
      h0f[idx] = v;
      h0bf[idx] = f2bf(v);
    }
  }
}

// ---------------- edge messages + scatter-add ----------------

__global__ void __launch_bounds__(256)
k_msg(const uint16_t* __restrict__ h0bf, const uint16_t* __restrict__ WmT,
      const float* __restrict__ bm,
      const int* __restrict__ esrc, const int* __restrict__ edst,
      float* __restrict__ pooled) {
  __shared__ uint16_t sW[128 * 256];          // 64 KB
  stage_to_lds(sW, WmT, 128 * 256 * 2);

  const int wave = (blockIdx.x * blockDim.x + threadIdx.x) >> 5;
  const int lane = threadIdx.x & 31;
  const int ntiles = NE / 16;
  if (wave >= ntiles) return;
  const int e0 = wave * 16;
  const int hi = lane >> 4;
  const int ln = lane & 15;

  const int sidx = esrc[e0 + ln];
  const int didx = edst[e0 + ln];
  const uint16_t* srow = h0bf + (size_t)sidx * DH;
  const uint16_t* drow = h0bf + (size_t)didx * DH;
  v16bf as0 = load_afrag(srow, 0, hi), as1 = load_afrag(srow, 32, hi);
  v16bf as2 = load_afrag(srow, 64, hi), as3 = load_afrag(srow, 96, hi);
  v16bf ad0 = load_afrag(drow, 0, hi), ad1 = load_afrag(drow, 32, hi);
  v16bf ad2 = load_afrag(drow, 64, hi), ad3 = load_afrag(drow, 96, hi);

  int drows[8];
#pragma unroll
  for (int r = 0; r < 8; ++r) drows[r] = edst[e0 + r + hi * 8];

  for (int ct = 0; ct < 8; ++ct) {
    const int col0 = ct * 16;
    const uint16_t* brow = sW + (size_t)(col0 + ln) * 256;  // LDS, [n][k0..255]
    float bias = bm[col0 + ln];
    v8f c = {bias, bias, bias, bias, bias, bias, bias, bias};
    c = WMMA_BF16(as0, load_bfrag(brow, 0, hi), c);
    c = WMMA_BF16(as1, load_bfrag(brow, 32, hi), c);
    c = WMMA_BF16(as2, load_bfrag(brow, 64, hi), c);
    c = WMMA_BF16(as3, load_bfrag(brow, 96, hi), c);
    c = WMMA_BF16(ad0, load_bfrag(brow, 128, hi), c);
    c = WMMA_BF16(ad1, load_bfrag(brow, 160, hi), c);
    c = WMMA_BF16(ad2, load_bfrag(brow, 192, hi), c);
    c = WMMA_BF16(ad3, load_bfrag(brow, 224, hi), c);
#pragma unroll
    for (int r = 0; r < 8; ++r) {
      float v = c[r];
      v = v > 0.0f ? v : 0.0f;                // relu
      atomicAdd(pooled + (size_t)drows[r] * DH + col0 + ln, v);
    }
  }
}

__global__ void k_cnt(const int* __restrict__ edst, float* __restrict__ cnt) {
  int i = blockIdx.x * blockDim.x + threadIdx.x;
  if (i < NE) atomicAdd(cnt + edst[i], 1.0f);
}

// ---------------- update + layernorm + graph pool ----------------

__device__ __forceinline__ v16bf make_pfrag(const float* prow, float invc, int kb, int hi) {
  union { uint16_t h[16]; v16bf v; } f;
  const float* p0 = prow + kb + hi * 8;
  const float* p1 = prow + kb + 16 + hi * 8;
#pragma unroll
  for (int j = 0; j < 8; ++j) f.h[j] = f2bf(p0[j] * invc);
#pragma unroll
  for (int j = 0; j < 8; ++j) f.h[8 + j] = f2bf(p1[j] * invc);
  return f.v;
}

__global__ void __launch_bounds__(256)
k_update(const uint16_t* __restrict__ h0bf, const float* __restrict__ h0f,
         const float* __restrict__ pooled, const float* __restrict__ cnt,
         const uint16_t* __restrict__ WuT, const float* __restrict__ bu,
         const float* __restrict__ gamma, const float* __restrict__ beta,
         float* __restrict__ gsum) {
  __shared__ uint16_t sW[128 * 256];          // 64 KB
  stage_to_lds(sW, WuT, 128 * 256 * 2);

  const int wave = (blockIdx.x * blockDim.x + threadIdx.x) >> 5;
  const int lane = threadIdx.x & 31;
  const int ntiles = NN / 16;
  if (wave >= ntiles) return;
  const int row0 = wave * 16;
  const int hi = lane >> 4;
  const int ln = lane & 15;

  const uint16_t* arow = h0bf + (size_t)(row0 + ln) * DH;
  v16bf a0 = load_afrag(arow, 0, hi), a1 = load_afrag(arow, 32, hi);
  v16bf a2 = load_afrag(arow, 64, hi), a3 = load_afrag(arow, 96, hi);

  const float invc = 1.0f / fmaxf(cnt[row0 + ln], 1.0f);
  const float* prow = pooled + (size_t)(row0 + ln) * DH;
  v16bf p0 = make_pfrag(prow, invc, 0, hi), p1 = make_pfrag(prow, invc, 32, hi);
  v16bf p2 = make_pfrag(prow, invc, 64, hi), p3 = make_pfrag(prow, invc, 96, hi);

  v8f acc[8];
#pragma unroll
  for (int ct = 0; ct < 8; ++ct) {
    const int col0 = ct * 16;
    const uint16_t* brow = sW + (size_t)(col0 + ln) * 256;  // LDS
    float bias = bu[col0 + ln];
    v8f c = {bias, bias, bias, bias, bias, bias, bias, bias};
    c = WMMA_BF16(a0, load_bfrag(brow, 0, hi), c);
    c = WMMA_BF16(a1, load_bfrag(brow, 32, hi), c);
    c = WMMA_BF16(a2, load_bfrag(brow, 64, hi), c);
    c = WMMA_BF16(a3, load_bfrag(brow, 96, hi), c);
    c = WMMA_BF16(p0, load_bfrag(brow, 128, hi), c);
    c = WMMA_BF16(p1, load_bfrag(brow, 160, hi), c);
    c = WMMA_BF16(p2, load_bfrag(brow, 192, hi), c);
    c = WMMA_BF16(p3, load_bfrag(brow, 224, hi), c);
    acc[ct] = c;
  }

  // residual + relu
#pragma unroll
  for (int ct = 0; ct < 8; ++ct) {
#pragma unroll
    for (int r = 0; r < 8; ++r) {
      int m = r + hi * 8;
      float v = acc[ct][r] + h0f[(size_t)(row0 + m) * DH + ct * 16 + ln];
      acc[ct][r] = v > 0.0f ? v : 0.0f;
    }
  }

  // layernorm stats per row: reduce 8 col-tiles, then across the 16 lanes of the row
  float mus[8], rss[8];
#pragma unroll
  for (int r = 0; r < 8; ++r) {
    float s = 0.0f, q = 0.0f;
#pragma unroll
    for (int ct = 0; ct < 8; ++ct) { float v = acc[ct][r]; s += v; q += v * v; }
#pragma unroll
    for (int off = 1; off < 16; off <<= 1) {
      s += __shfl_xor(s, off, 32);
      q += __shfl_xor(q, off, 32);
    }
    float mu = s * (1.0f / 128.0f);
    float var = q * (1.0f / 128.0f) - mu * mu;
    mus[r] = mu;
    rss[r] = rsqrtf(var + 1e-6f);
  }

  // normalize, scale/shift, accumulate per-column partial for the graph mean
#pragma unroll
  for (int ct = 0; ct < 8; ++ct) {
    float g = gamma[ct * 16 + ln];
    float bt = beta[ct * 16 + ln];
    float colp = 0.0f;
#pragma unroll
    for (int r = 0; r < 8; ++r) {
      float v = (acc[ct][r] - mus[r]) * rss[r] * g + bt;
      colp += v;
    }
    colp += __shfl_xor(colp, 16, 32);   // rows 0-7 + rows 8-15, same column
    if (hi == 0) atomicAdd(gsum + ct * 16 + ln, colp);
  }
}

// ---------------- final [1,128] @ [128,32] ----------------

__global__ void k_out(const float* __restrict__ gsum, const float* __restrict__ Wd,
                      const float* __restrict__ bd, float* __restrict__ out) {
  int o = threadIdx.x;
  if (o < OUTD) {
    float s = bd[o];
    const float inv_n = 1.0f / (float)NN;
#pragma unroll 4
    for (int k = 0; k < DH; ++k) s += gsum[k] * inv_n * Wd[k * OUTD + o];
    out[o] = s;
  }
}

// ---------------- launch ----------------

extern "C" void kernel_launch(void* const* d_in, const int* in_sizes, int n_in,
                              void* d_out, int out_size, void* d_ws, size_t ws_size,
                              hipStream_t stream) {
  const float* x     = (const float*)d_in[0];
  const int*   esrc  = (const int*)d_in[1];
  const int*   edst  = (const int*)d_in[2];
  const float* Wp    = (const float*)d_in[3];
  const float* bp    = (const float*)d_in[4];
  const float* Wm    = (const float*)d_in[5];
  const float* bm    = (const float*)d_in[6];
  const float* Wu    = (const float*)d_in[7];
  const float* bu    = (const float*)d_in[8];
  const float* gamma = (const float*)d_in[9];
  const float* beta  = (const float*)d_in[10];
  const float* Wd    = (const float*)d_in[11];
  const float* bd    = (const float*)d_in[12];
  float* out = (float*)d_out;

  // workspace layout (256B aligned slabs)
  char* ws = (char*)d_ws;
  size_t off = 0;
  auto take = [&](size_t bytes) { char* p = ws + off; off = (off + bytes + 255) & ~(size_t)255; return p; };
  uint16_t* xbf    = (uint16_t*)take((size_t)NN * DH * 2);
  uint16_t* WpT    = (uint16_t*)take(128 * 128 * 2);
  uint16_t* WmT    = (uint16_t*)take(128 * 256 * 2);
  uint16_t* WuT    = (uint16_t*)take(128 * 256 * 2);
  float*    h0f    = (float*)take((size_t)NN * DH * 4);
  uint16_t* h0bf   = (uint16_t*)take((size_t)NN * DH * 2);
  float*    pooled = (float*)take((size_t)NN * DH * 4);
  float*    cnt    = (float*)take((size_t)NN * 4);
  float*    gsum   = (float*)take(DH * 4);
  (void)ws_size; (void)n_in; (void)in_sizes; (void)out_size;

  // zero accumulators every call (graph-capture-safe)
  (void)hipMemsetAsync(pooled, 0, (size_t)NN * DH * 4, stream);
  (void)hipMemsetAsync(cnt, 0, (size_t)NN * 4, stream);
  (void)hipMemsetAsync(gsum, 0, DH * 4, stream);

  // prep
  k_cvt_x<<<(NN * DH + 255) / 256, 256, 0, stream>>>(x, xbf, NN * DH);
  k_transpose_w<<<(128 * 128 + 255) / 256, 256, 0, stream>>>(Wp, WpT, 128, 128);
  k_transpose_w<<<(256 * 128 + 255) / 256, 256, 0, stream>>>(Wm, WmT, 256, 128);
  k_transpose_w<<<(256 * 128 + 255) / 256, 256, 0, stream>>>(Wu, WuT, 256, 128);

  // h0
  {
    int tiles = NN / 16;                       // 3125
    int blocks = (tiles + 7) / 8;              // 8 waves/block
    k_proj<<<blocks, 256, 0, stream>>>(xbf, WpT, bp, h0f, h0bf);
  }
  // degrees
  k_cnt<<<(NE + 255) / 256, 256, 0, stream>>>(edst, cnt);
  // messages + scatter
  {
    int tiles = NE / 16;                       // 40000
    int blocks = (tiles + 7) / 8;              // 5000
    k_msg<<<blocks, 256, 0, stream>>>(h0bf, WmT, bm, esrc, edst, pooled);
  }
  // update + LN + graph pool
  {
    int tiles = NN / 16;
    int blocks = (tiles + 7) / 8;
    k_update<<<blocks, 256, 0, stream>>>(h0bf, h0f, pooled, cnt, WuT, bu, gamma, beta, gsum);
  }
  // readout
  k_out<<<1, 32, 0, stream>>>(gsum, Wd, bd, out);
}
